// PEFTTransformerTradingAgent_55370718380257
// MI455X (gfx1250) — matrix operand order repaired and checked by reading, not compiled, not run on hardware
//
#include <hip/hip_runtime.h>
#include <hip/hip_bf16.h>
#include <math.h>

// ---------------- problem constants ----------------
#define BB   16
#define SS   1024
#define DINP 64
#define HID  512
#define NHEADS 8
#define HD   64
#define FFD  2048
#define MTOT (BB*SS)   // 16384 token rows

typedef __attribute__((ext_vector_type(16))) _Float16 v16h;
typedef __attribute__((ext_vector_type(8)))  _Float16 v8h;
typedef __attribute__((ext_vector_type(4)))  _Float16 v4h;
typedef __attribute__((ext_vector_type(2)))  _Float16 v2h;
typedef __attribute__((ext_vector_type(8)))  float    v8f;

#define WMMA_F32_F16(a,b,c) \
  __builtin_amdgcn_wmma_f32_16x16x32_f16(false,(a),false,(b),(short)0,(c),false,false)

// ---------- WMMA fragment loaders (CDNA5 VGPR layouts, wave32) ----------
// A-matrix 16x32 f16: lanes0-15 row=lane: K = e<8 ? e : 8+e ; lanes16-31: +8 / +16.
__device__ __forceinline__ v16h frag_a(const _Float16* p16, int stride, int row_base) {
  const int lane = threadIdx.x & 31;
  const int hs = lane >> 4;
  const _Float16* p = p16 + (row_base + (lane & 15)) * stride;
  v8h lo = *(const v8h*)(p + (hs ? 8 : 0));
  v8h hi = *(const v8h*)(p + (hs ? 24 : 16));
  return __builtin_shufflevector(lo, hi, 0,1,2,3,4,5,6,7,8,9,10,11,12,13,14,15);
}
// B-matrix 32x16 f16: lane n=lane&15, K = e<8 ? e : 8+e ; lanes16-31: +8 / +16 (same as A).
__device__ __forceinline__ v16h frag_b(const _Float16* p16, int stride, int row_base) {
  const int lane = threadIdx.x & 31;
  const _Float16* p = p16 + (row_base + (lane & 15)) * stride + ((lane >> 4) << 3);
  v8h lo = *(const v8h*)(p);
  v8h hi = *(const v8h*)(p + 16);
  return __builtin_shufflevector(lo, hi, 0,1,2,3,4,5,6,7,8,9,10,11,12,13,14,15);
}

__device__ __forceinline__ v4h cvt4(float4 v) {
  v4h h; h[0] = (_Float16)v.x; h[1] = (_Float16)v.y;
         h[2] = (_Float16)v.z; h[3] = (_Float16)v.w;
  return h;
}

// ---------------- LoRA merge: W' = f16(W + s * B@A) ----------------
__global__ __launch_bounds__(256)
void merge_lora(const float* __restrict__ W, const float* __restrict__ A,
                const float* __restrict__ Bm, _Float16* __restrict__ out,
                int dout, int din, int rank, float scale) {
  int i = blockIdx.x * blockDim.x + threadIdx.x;
  if (i >= dout * din) return;
  int n = i / din, k = i - n * din;
  float acc = W[i];
  for (int r = 0; r < rank; ++r) acc += scale * Bm[n * rank + r] * A[r * din + k];
  out[i] = (_Float16)acc;
}

// -------- double-buffered WMMA GEMM: Y = act(X @ W'^T + bias [+ pe]) --------
// X: [M,K] f32 (row stride K), W': [N,K] f16, Y: [M,N] f32.  64x64x32 tiles.
// act: 0=none 1=gelu(exact) 2=relu 3=add positional encoding
template<bool BOUNDED>
__global__ __launch_bounds__(128)
void gemm_wmma(const float* __restrict__ X, const _Float16* __restrict__ W,
               const float* __restrict__ bias, float* __restrict__ Y,
               int M, int N, int K, int act, const float* __restrict__ pe) {
  __shared__ _Float16 Xs[2][64 * 40];   // ping-pong, pad 32->40 halves
  __shared__ _Float16 Ws[2][64 * 40];
  const int tid  = threadIdx.x;
  const int wave = tid >> 5;
  const int wr = (wave >> 1) * 32;      // wave row offset in 64x64 tile
  const int wc = (wave & 1)  * 32;      // wave col offset
  const int tm = blockIdx.x * 64, tn = blockIdx.y * 64;

  const int rx = tid >> 3, cx = (tid & 7) * 4;  // X: rows rx+{0,16,32,48}, float4 at cx
  const int rw = tid >> 2, cw = (tid & 3) * 8;  // W: rows rw+{0,32}, v8h at cw

  float4 xg[4]; v8h wg[2];
  auto load_g = [&](int k0) {
#pragma unroll
    for (int j = 0; j < 4; ++j) {
      int gm = tm + rx + j * 16;
      if (BOUNDED && gm >= M) xg[j] = make_float4(0.f, 0.f, 0.f, 0.f);
      else                    xg[j] = *(const float4*)(X + (size_t)gm * K + k0 + cx);
    }
#pragma unroll
    for (int j = 0; j < 2; ++j)
      wg[j] = *(const v8h*)(W + (size_t)(tn + rw + j * 32) * K + k0 + cw);
  };
  auto store_l = [&](int buf) {
#pragma unroll
    for (int j = 0; j < 4; ++j)
      *(v4h*)(&Xs[buf][(rx + j * 16) * 40 + cx]) = cvt4(xg[j]);
#pragma unroll
    for (int j = 0; j < 2; ++j)
      *(v8h*)(&Ws[buf][(rw + j * 32) * 40 + cw]) = wg[j];
  };

  v8f acc[2][2] = {};
  load_g(0);
  store_l(0);
  __syncthreads();

  const int nk = K >> 5;
  for (int t = 0; t < nk; ++t) {
    const int cur = t & 1;
    if (t + 1 < nk) {
      load_g((t + 1) << 5);                       // issue next tile's global loads
      if (!BOUNDED && t + 2 < nk)                 // prefetch 2 tiles ahead
        __builtin_prefetch(X + (size_t)(tm + rx) * K + ((t + 2) << 5) + cx, 0, 0);
    }
    v16h a0 = frag_a(Xs[cur], 40, wr);
    v16h a1 = frag_a(Xs[cur], 40, wr + 16);
    v16h b0 = frag_b(Ws[cur], 40, wc);
    v16h b1 = frag_b(Ws[cur], 40, wc + 16);
    acc[0][0] = WMMA_F32_F16(a0, b0, acc[0][0]);
    acc[0][1] = WMMA_F32_F16(a0, b1, acc[0][1]);
    acc[1][0] = WMMA_F32_F16(a1, b0, acc[1][0]);
    acc[1][1] = WMMA_F32_F16(a1, b1, acc[1][1]);
    if (t + 1 < nk) store_l(cur ^ 1);
    __syncthreads();
  }

  const int lane = tid & 31;
  const int hs = lane >> 4, nl = lane & 15;
#pragma unroll
  for (int mi = 0; mi < 2; ++mi)
#pragma unroll
    for (int ni = 0; ni < 2; ++ni) {
      int gn = tn + wc + ni * 16 + nl;
      float bv = bias ? bias[gn] : 0.f;
#pragma unroll
      for (int r = 0; r < 8; ++r) {
        int gm = tm + wr + mi * 16 + 8 * hs + r;
        if (BOUNDED && gm >= M) continue;
        float v = acc[mi][ni][r] + bv;
        if      (act == 1) v = 0.5f * v * (1.f + erff(v * 0.70710678118f));
        else if (act == 2) v = fmaxf(v, 0.f);
        else if (act == 3) v += pe[(size_t)(gm & (SS - 1)) * HID + gn];
        Y[(size_t)gm * N + gn] = v;
      }
    }
}

// ------ flash-style attention: 4 waves/block share K/V staging, 64 queries ------
// Row sums computed by an extra WMMA against an all-ones B fragment (no shuffles);
// running max reduced with packed-f16 lane shuffles (stabilization only).
__global__ __launch_bounds__(128)
void attn_fwd(const float* __restrict__ Qm, const float* __restrict__ Km,
              const float* __restrict__ Vm, float* __restrict__ Om) {
  __shared__ _Float16 Qs[64 * 72];      // [q][d]
  __shared__ _Float16 Ks[32 * 72];      // [key][d]
  __shared__ _Float16 Vt[64 * 40];      // [d][key] transposed for B-frags
  __shared__ _Float16 Ps[4][16 * 40];   // per-wave probs (C-layout -> A-layout bounce)
  const int tid = threadIdx.x;
  const int wave = tid >> 5, lane = tid & 31;
  const int bh = blockIdx.y, b = bh >> 3, h = bh & 7;
  const int q0 = blockIdx.x * 64;
  const size_t base = (size_t)b * SS * HID + (size_t)h * HD;

  // stage Q 64x64 cooperatively
  {
    const int r = tid >> 1, c0 = (tid & 1) * 32;
#pragma unroll
    for (int j = 0; j < 8; ++j) {
      int c = c0 + j * 4;
      float4 v = *(const float4*)(Qm + base + (size_t)(q0 + r) * HID + c);
      *(v4h*)(&Qs[r * 72 + c]) = cvt4(v);
    }
  }
  __syncthreads();
  v16h qa0 = frag_a(Qs,      72, wave * 16);   // d 0..31
  v16h qa1 = frag_a(Qs + 32, 72, wave * 16);   // d 32..63

  v16h onesf;                                   // all-ones B fragment (row-sum trick)
#pragma unroll
  for (int i = 0; i < 16; ++i) onesf[i] = (_Float16)1.f;

  v8f o[4] = {};
  v8f osum = {};                                // per-row sums (replicated across lanes)
  float rowM[8];
#pragma unroll
  for (int r = 0; r < 8; ++r) rowM[r] = -3.0e38f;

  // staging maps
  const int rk = tid >> 2, ck = (tid & 3) * 16;     // K: row rk, 4x float4
  const int rp2 = (tid >> 3) * 2, cv = (tid & 7) * 8; // V: row pair {rp2,rp2+1}, 8 cols

  for (int kc = 0; kc < SS; kc += 32) {
    __syncthreads();
    // stage K naturally (vectorized v4h)
#pragma unroll
    for (int j = 0; j < 4; ++j) {
      int c = ck + j * 4;
      float4 kv = *(const float4*)(Km + base + (size_t)(kc + rk) * HID + c);
      *(v4h*)(&Ks[rk * 72 + c]) = cvt4(kv);
    }
    // stage V transposed: adjacent key pair packs into one b32 store per d
    {
      float4 e0 = *(const float4*)(Vm + base + (size_t)(kc + rp2) * HID + cv);
      float4 e1 = *(const float4*)(Vm + base + (size_t)(kc + rp2) * HID + cv + 4);
      float4 d0 = *(const float4*)(Vm + base + (size_t)(kc + rp2 + 1) * HID + cv);
      float4 d1 = *(const float4*)(Vm + base + (size_t)(kc + rp2 + 1) * HID + cv + 4);
      auto stp = [&](int dcol, float ev, float ov) {
        v2h p; p[0] = (_Float16)ev; p[1] = (_Float16)ov;
        *(v2h*)(&Vt[(cv + dcol) * 40 + rp2]) = p;
      };
      stp(0, e0.x, d0.x); stp(1, e0.y, d0.y); stp(2, e0.z, d0.z); stp(3, e0.w, d0.w);
      stp(4, e1.x, d1.x); stp(5, e1.y, d1.y); stp(6, e1.z, d1.z); stp(7, e1.w, d1.w);
    }
    __syncthreads();

    // scores: two 16x16 tiles over this 32-key chunk (d=64 -> 2 WMMAs each)
    v8f s0 = {}, s1 = {};
    {
      v16h kb0 = frag_b(Ks,      72, 0);
      v16h kb1 = frag_b(Ks + 32, 72, 0);
      s0 = WMMA_F32_F16(qa0, kb0, s0);
      s0 = WMMA_F32_F16(qa1, kb1, s0);
      v16h kb2 = frag_b(Ks,      72, 16);
      v16h kb3 = frag_b(Ks + 32, 72, 16);
      s1 = WMMA_F32_F16(qa0, kb2, s1);
      s1 = WMMA_F32_F16(qa1, kb3, s1);
    }

    // chunk row-max, packed f16 (4 regs x 4 xor steps), then online rescale
    v2h mpk[4];
#pragma unroll
    for (int j = 0; j < 4; ++j) {
      mpk[j][0] = (_Float16)(fmaxf(s0[2*j],   s1[2*j])   * 0.125f);
      mpk[j][1] = (_Float16)(fmaxf(s0[2*j+1], s1[2*j+1]) * 0.125f);
    }
#pragma unroll
    for (int mask = 1; mask <= 8; mask <<= 1)
#pragma unroll
      for (int j = 0; j < 4; ++j) {
        int other = __shfl_xor(__builtin_bit_cast(int, mpk[j]), mask);
        mpk[j] = __builtin_elementwise_max(mpk[j], __builtin_bit_cast(v2h, other));
      }

    float p0[8], p1[8], alpha[8];
#pragma unroll
    for (int r = 0; r < 8; ++r) {
      float nm = fmaxf(rowM[r], (float)mpk[r >> 1][r & 1]);
      alpha[r] = __expf(rowM[r] - nm);
      p0[r] = __expf(s0[r] * 0.125f - nm);
      p1[r] = __expf(s1[r] * 0.125f - nm);
      rowM[r] = nm;
    }
#pragma unroll
    for (int t = 0; t < 4; ++t)
#pragma unroll
      for (int r = 0; r < 8; ++r) o[t][r] *= alpha[r];
#pragma unroll
    for (int r = 0; r < 8; ++r) osum[r] *= alpha[r];

    // C-layout -> A-layout bounce through per-wave LDS scratch
    {
      int hs = lane >> 4, nl = lane & 15;
#pragma unroll
      for (int r = 0; r < 8; ++r) {
        int m = r + 8 * hs;
        Ps[wave][m * 40 + nl]      = (_Float16)p0[r];
        Ps[wave][m * 40 + 16 + nl] = (_Float16)p1[r];
      }
    }
    __syncthreads();
    v16h pa = frag_a(Ps[wave], 40, 0);
    osum = WMMA_F32_F16(pa, onesf, osum);       // row sums, replicated to all lanes
#pragma unroll
    for (int t = 0; t < 4; ++t) {
      v16h vb = frag_b(Vt, 40, t * 16);
      o[t] = WMMA_F32_F16(pa, vb, o[t]);
    }
  }

  // normalize + store ctx
  const int hs = lane >> 4, nl = lane & 15;
#pragma unroll
  for (int r = 0; r < 8; ++r) {
    float inv = 1.0f / osum[r];
    int m = q0 + wave * 16 + r + 8 * hs;
#pragma unroll
    for (int t = 0; t < 4; ++t)
      Om[base + (size_t)m * HID + t * 16 + nl] = o[t][r] * inv;
  }
}

// ---------------- residual add + LayerNorm (one block per row, HID=512) ----------------
__global__ __launch_bounds__(128)
void add_ln(const float* __restrict__ x, const float* __restrict__ res,
            const float* __restrict__ g, const float* __restrict__ bb,
            float* __restrict__ y) {
  __shared__ float s1[128], s2[128];
  const int row = blockIdx.x, t = threadIdx.x;
  const float* xr = x + (size_t)row * HID;
  const float* rr = res ? res + (size_t)row * HID : nullptr;
  float v[4]; float sum = 0.f, sq = 0.f;
#pragma unroll
  for (int j = 0; j < 4; ++j) {
    float a = xr[t + j * 128] + (rr ? rr[t + j * 128] : 0.f);
    v[j] = a; sum += a; sq += a * a;
  }
  s1[t] = sum; s2[t] = sq; __syncthreads();
  for (int o = 64; o > 0; o >>= 1) {
    if (t < o) { s1[t] += s1[t + o]; s2[t] += s2[t + o]; }
    __syncthreads();
  }
  float mean = s1[0] * (1.0f / HID);
  float var  = s2[0] * (1.0f / HID) - mean * mean;
  float inv  = rsqrtf(var + 1e-5f);
#pragma unroll
  for (int j = 0; j < 4; ++j) {
    int c = t + j * 128;
    y[(size_t)row * HID + c] = (v[j] - mean) * inv * g[c] + bb[c];
  }
}

// ---------------- mean over sequence ----------------
__global__ __launch_bounds__(256)
void seq_mean(const float* __restrict__ hbuf, float* __restrict__ f) {
  int idx = blockIdx.x * blockDim.x + threadIdx.x;
  if (idx >= BB * HID) return;
  int b = idx >> 9, c = idx & (HID - 1);
  float s = 0.f;
  for (int t = 0; t < SS; ++t) s += hbuf[((size_t)b * SS + t) * HID + c];
  f[idx] = s * (1.0f / SS);
}

// ---------------- final head output (dot64 + optional tanh) ----------------
__global__ __launch_bounds__(32)
void head_out(const float* __restrict__ h2, const float* __restrict__ W,
              const float* __restrict__ bias, float* __restrict__ out, int do_tanh) {
  int b = threadIdx.x;
  if (b >= BB) return;
  float acc = bias[0];
  for (int k = 0; k < 64; ++k) acc += h2[b * 64 + k] * W[k];
  out[b] = do_tanh ? tanhf(acc) : acc;
}

// =======================================================================
extern "C" void kernel_launch(void* const* d_in, const int* in_sizes, int n_in,
                              void* d_out, int out_size, void* d_ws, size_t ws_size,
                              hipStream_t stream) {
  (void)in_sizes; (void)n_in; (void)out_size; (void)ws_size;
  auto FP = [&](int i) -> const float* { return (const float*)d_in[i]; };

  char* ws = (char*)d_ws;
  size_t off = 0;
  auto alloc = [&](size_t bytes) -> char* {
    char* p = ws + off; off += (bytes + 255) & ~(size_t)255; return p;
  };

  // merged f16 weights
  _Float16* wh_in = (_Float16*)alloc((size_t)HID * DINP * 2);
  _Float16 *wh_q[4], *wh_k[4], *wh_v[4], *wh_o[4], *wh_f1[4], *wh_f2[4];
  for (int l = 0; l < 4; ++l) {
    wh_q[l]  = (_Float16*)alloc((size_t)HID * HID * 2);
    wh_k[l]  = (_Float16*)alloc((size_t)HID * HID * 2);
    wh_v[l]  = (_Float16*)alloc((size_t)HID * HID * 2);
    wh_o[l]  = (_Float16*)alloc((size_t)HID * HID * 2);
    wh_f1[l] = (_Float16*)alloc((size_t)FFD * HID * 2);
    wh_f2[l] = (_Float16*)alloc((size_t)HID * FFD * 2);
  }
  _Float16* wh_a1 = (_Float16*)alloc((size_t)128 * 512 * 2);
  _Float16* wh_a2 = (_Float16*)alloc((size_t)64 * 128 * 2);
  _Float16* wh_c1 = (_Float16*)alloc((size_t)128 * 512 * 2);
  _Float16* wh_c2 = (_Float16*)alloc((size_t)64 * 128 * 2);

  // activation buffers (f32)
  float* H   = (float*)alloc((size_t)MTOT * HID * 4);
  float* Tq  = (float*)alloc((size_t)MTOT * HID * 4);
  float* Tk  = (float*)alloc((size_t)MTOT * HID * 4);
  float* Tv  = (float*)alloc((size_t)MTOT * HID * 4);
  float* Tc  = (float*)alloc((size_t)MTOT * HID * 4);
  float* FFB = (float*)alloc((size_t)MTOT * FFD * 4);
  float* Fm  = (float*)alloc((size_t)BB * HID * 4);
  float* H1  = (float*)alloc((size_t)BB * 128 * 4);
  float* H2  = (float*)alloc((size_t)BB * 64 * 4);

  auto merge = [&](const float* W, const float* A, const float* Bm,
                   _Float16* outp, int dout, int din, int rank) {
    int n = dout * din;
    merge_lora<<<dim3((n + 255) / 256), dim3(256), 0, stream>>>(
        W, A, Bm, outp, dout, din, rank, 2.0f);
  };
  auto gemm = [&](const float* X, const _Float16* W, const float* bias, float* Y,
                  int M, int N, int K, int act, const float* pe) {
    dim3 g((M + 63) / 64, N / 64);
    if ((M & 63) == 0)
      gemm_wmma<false><<<g, dim3(128), 0, stream>>>(X, W, bias, Y, M, N, K, act, pe);
    else
      gemm_wmma<true><<<g, dim3(128), 0, stream>>>(X, W, bias, Y, M, N, K, act, pe);
  };

  // ---- input index map: setup_inputs() dict insertion order ----
  // 0:x  1-4:in_proj{W,b,A,B}
  // layer l base LB=5+28l: q{W,b,A,B} k v o ff1 ff2 ln1{g,b} ln2{g,b}
  // 117,118: ln_f{g,b}
  // actor 119-128: l1{W,b,A,B} l2{W,b,A,B} out_W out_b ; critic 129-138 ; 139: pe
  merge(FP(1), FP(3), FP(4), wh_in, HID, DINP, 8);
  for (int l = 0; l < 4; ++l) {
    int LB = 5 + l * 28;
    merge(FP(LB + 0),  FP(LB + 2),  FP(LB + 3),  wh_q[l],  HID, HID, 8);
    merge(FP(LB + 4),  FP(LB + 6),  FP(LB + 7),  wh_k[l],  HID, HID, 8);
    merge(FP(LB + 8),  FP(LB + 10), FP(LB + 11), wh_v[l],  HID, HID, 8);
    merge(FP(LB + 12), FP(LB + 14), FP(LB + 15), wh_o[l],  HID, HID, 8);
    merge(FP(LB + 16), FP(LB + 18), FP(LB + 19), wh_f1[l], FFD, HID, 8);
    merge(FP(LB + 20), FP(LB + 22), FP(LB + 23), wh_f2[l], HID, FFD, 8);
  }
  merge(FP(119), FP(121), FP(122), wh_a1, 128, 512, 4);
  merge(FP(123), FP(125), FP(126), wh_a2, 64, 128, 2);
  merge(FP(129), FP(131), FP(132), wh_c1, 128, 512, 4);
  merge(FP(133), FP(135), FP(136), wh_c2, 64, 128, 2);

  const float* pe = FP(139);

  // in_proj + positional encoding
  gemm(FP(0), wh_in, FP(2), H, MTOT, HID, DINP, 3, pe);

  for (int l = 0; l < 4; ++l) {
    int LB = 5 + l * 28;
    gemm(H, wh_q[l], FP(LB + 1), Tq, MTOT, HID, HID, 0, nullptr);
    gemm(H, wh_k[l], FP(LB + 5), Tk, MTOT, HID, HID, 0, nullptr);
    gemm(H, wh_v[l], FP(LB + 9), Tv, MTOT, HID, HID, 0, nullptr);
    attn_fwd<<<dim3(SS / 64, BB * NHEADS), dim3(128), 0, stream>>>(Tq, Tk, Tv, Tc);
    gemm(Tc, wh_o[l], FP(LB + 13), Tq, MTOT, HID, HID, 0, nullptr);   // reuse Tq
    add_ln<<<dim3(MTOT), dim3(128), 0, stream>>>(H, Tq, FP(LB + 24), FP(LB + 25), H);
    gemm(H, wh_f1[l], FP(LB + 17), FFB, MTOT, FFD, HID, 1, nullptr);  // GELU
    gemm(FFB, wh_f2[l], FP(LB + 21), Tq, MTOT, HID, FFD, 0, nullptr);
    add_ln<<<dim3(MTOT), dim3(128), 0, stream>>>(H, Tq, FP(LB + 26), FP(LB + 27), H);
  }

  // final LN, sequence mean
  add_ln<<<dim3(MTOT), dim3(128), 0, stream>>>(H, nullptr, FP(117), FP(118), H);
  seq_mean<<<dim3((BB * HID + 255) / 256), dim3(256), 0, stream>>>(H, Fm);

  float* outp = (float*)d_out;
  // actor head -> out[0:16]
  gemm(Fm, wh_a1, FP(120), H1, BB, 128, 512, 2, nullptr);
  gemm(H1, wh_a2, FP(124), H2, BB, 64, 128, 2, nullptr);
  head_out<<<dim3(1), dim3(32), 0, stream>>>(H2, FP(127), FP(128), outp, 1);
  // critic head -> out[16:32]
  gemm(Fm, wh_c1, FP(130), H1, BB, 128, 512, 2, nullptr);
  gemm(H1, wh_c2, FP(134), H2, BB, 64, 128, 2, nullptr);
  head_out<<<dim3(1), dim3(32), 0, stream>>>(H2, FP(137), FP(138), outp + BB, 0);
}
